// DecoderRNN_90537910600223
// MI455X (gfx1250) — matrix operand order repaired and checked by reading, compiled, not run on hardware
//
#include <hip/hip_runtime.h>
#include <hip/hip_bf16.h>

typedef __attribute__((ext_vector_type(16))) _Float16 v16h;
typedef __attribute__((ext_vector_type(8)))  _Float16 v8h;
typedef __attribute__((ext_vector_type(8)))  float    v8f;

#define BB   128
#define NN   196
#define DV   512
#define EE   512
#define HH   1024
#define VV   10000
#define TT   20
#define XK   2048   // x row = [ctx(512) | emb(512) | h(1024)]
#define G4H  4096   // 4*H

#define MT   2      // m-tiles per wave (32 rows)
#define NT   4      // n-tiles per wave (64 cols)

// ---------------------------------------------------------------------------
// Register-tiled WMMA GEMM:
//   C[M x Nout] (f32) = A[M x K] (f16, row-major, lda) * Wt[Nout x K]^T (+bias)
// Each wave owns a 32x64 output tile (MT x NT 16x16 WMMA tiles):
//   per K-step-32: 4 A-loads + 8 B-loads (b128) feed 8 v_wmma ops.
// 4 waves per block. N padding handled by clamped loads + guarded stores
// (guards are wave-uniform -> EXEC all-ones for every WMMA).
// ---------------------------------------------------------------------------
__global__ void wmma_gemm_tiled(const _Float16* __restrict__ A, int lda,
                                const _Float16* __restrict__ Wt, int ldw,
                                float* __restrict__ C, int ldc,
                                const float* __restrict__ bias,
                                int M, int Nout, int K)
{
    const int lane   = threadIdx.x & 31;
    const int wave   = threadIdx.x >> 5;
    const int Ntiles = (Nout + 15) >> 4;
    const int ntile0 = (blockIdx.x * 4 + wave) * NT;
    const int mtile0 = blockIdx.y * MT;
    if (ntile0 >= Ntiles) return;            // whole-wave exit

    const int r    = lane & 15;
    const int half = lane >> 4;

    const _Float16* aptr[MT];
#pragma unroll
    for (int i = 0; i < MT; ++i)
        aptr[i] = A + (size_t)((mtile0 + i) * 16 + r) * lda + half * 8;

    const _Float16* bptr[NT];
#pragma unroll
    for (int j = 0; j < NT; ++j) {
        int nt = ntile0 + j;
        if (nt > Ntiles - 1) nt = Ntiles - 1;          // clamp loads into range
        bptr[j] = Wt + (size_t)(nt * 16 + r) * ldw + half * 8;
    }

    v8f acc[MT][NT] = {};
    for (int k = 0; k < K; k += 32) {
        v16h af[MT], bf[NT];
#pragma unroll
        for (int i = 0; i < MT; ++i) {
            v8h lo = *(const v8h*)(aptr[i] + k);        // K = half*8 .. +7
            v8h hi = *(const v8h*)(aptr[i] + k + 16);   // K = 16+half*8 .. +7
#pragma unroll
            for (int q = 0; q < 8; ++q) { af[i][q] = lo[q]; af[i][q + 8] = hi[q]; }
        }
#pragma unroll
        for (int j = 0; j < NT; ++j) {
            v8h lo = *(const v8h*)(bptr[j] + k);        // K = half*8 .. +7
            v8h hi = *(const v8h*)(bptr[j] + k + 16);   // K = 16+half*8 .. +7
#pragma unroll
            for (int q = 0; q < 8; ++q) { bf[j][q] = lo[q]; bf[j][q + 8] = hi[q]; }
        }
#pragma unroll
        for (int i = 0; i < MT; ++i)
#pragma unroll
            for (int j = 0; j < NT; ++j)
                acc[i][j] = __builtin_amdgcn_wmma_f32_16x16x32_f16(
                    false, af[i], false, bf[j], (short)0, acc[i][j], false, false);
    }

    const int mrow = half * 8;
#pragma unroll
    for (int i = 0; i < MT; ++i) {
#pragma unroll
        for (int j = 0; j < NT; ++j) {
            const int nt = ntile0 + j;
            if (nt >= Ntiles) continue;                // wave-uniform guard
            const float bv = bias ? bias[nt * 16 + r] : 0.0f;
#pragma unroll
            for (int v = 0; v < 8; ++v)
                C[(size_t)((mtile0 + i) * 16 + mrow + v) * ldc + nt * 16 + r]
                    = acc[i][j][v] + bv;
        }
    }
}

// ---------------------------------------------------------------------------
// fp32 -> f16 convert
// ---------------------------------------------------------------------------
__global__ void cvt_f32_f16(const float* __restrict__ src,
                            _Float16* __restrict__ dst, int n)
{
    int i = blockIdx.x * blockDim.x + threadIdx.x;
    if (i < n) dst[i] = (_Float16)src[i];
}

// [W_ih | W_hh] row-concat into f16 (4096 x 2048)
__global__ void cvt_concat_wcat(const float* __restrict__ W_ih,
                                const float* __restrict__ W_hh,
                                _Float16* __restrict__ dst)
{
    int i = blockIdx.x * blockDim.x + threadIdx.x;   // over 4096*2048
    if (i >= G4H * XK) return;
    int row = i / XK, col = i % XK;
    float v = (col < 1024) ? W_ih[(size_t)row * 1024 + col]
                           : W_hh[(size_t)row * 1024 + (col - 1024)];
    dst[i] = (_Float16)v;
}

__global__ void bias_combine(const float* __restrict__ b_ih,
                             const float* __restrict__ b_hh,
                             float* __restrict__ dst)
{
    int i = blockIdx.x * blockDim.x + threadIdx.x;
    if (i < G4H) dst[i] = b_ih[i] + b_hh[i];
}

// mean over N of features -> f16 (B x Dv)
__global__ void featmean_kernel(const float* __restrict__ features,
                                _Float16* __restrict__ fm)
{
    int b = blockIdx.x;
    for (int d = threadIdx.x; d < DV; d += 256) {
        float acc = 0.0f;
        for (int n = 0; n < NN; ++n)
            acc += features[((size_t)b * NN + n) * DV + d];
        fm[(size_t)b * DV + d] = (_Float16)(acc / (float)NN);
    }
}

// att_v[b,n] = dot(features[b,n,:], W_attn_v) + b_attn_v
__global__ void attv_kernel(const float* __restrict__ features,
                            const float* __restrict__ W_attn_v,
                            const float* __restrict__ b_attn_v,
                            float* __restrict__ att_v)
{
    int bn = blockIdx.x;                 // B*N blocks
    __shared__ float red[128];
    float s = 0.0f;
    for (int k = threadIdx.x; k < DV; k += 128)
        s += features[(size_t)bn * DV + k] * W_attn_v[k];
    red[threadIdx.x] = s; __syncthreads();
    for (int o = 64; o > 0; o >>= 1) {
        if (threadIdx.x < o) red[threadIdx.x] += red[threadIdx.x + o];
        __syncthreads();
    }
    if (threadIdx.x == 0) att_v[bn] = red[0] + b_attn_v[0];
}

// fused: att_h dot, softmax over N (incl. att_h for fidelity), ctx -> x[:,0:512] f16
__global__ void attn_ctx_kernel(const float* __restrict__ features,
                                const float* __restrict__ att_v,
                                const float* __restrict__ hbuf,
                                const float* __restrict__ W_attn_h,
                                const float* __restrict__ b_attn_h,
                                _Float16* __restrict__ x)
{
    int b = blockIdx.x;
    int tid = threadIdx.x;
    __shared__ float red[256];
    __shared__ float alpha[NN];

    // att_h = h . W_attn_h + b
    float s = 0.0f;
    for (int k = tid; k < HH; k += 256) s += hbuf[(size_t)b * HH + k] * W_attn_h[k];
    red[tid] = s; __syncthreads();
    for (int o = 128; o > 0; o >>= 1) {
        if (tid < o) red[tid] += red[tid + o];
        __syncthreads();
    }
    float att_h = red[0] + b_attn_h[0];
    __syncthreads();

    // softmax over N
    float mx = -3.402823e38f;
    for (int n = tid; n < NN; n += 256) {
        float l = att_v[(size_t)b * NN + n] + att_h;
        alpha[n] = l;
        mx = fmaxf(mx, l);
    }
    red[tid] = mx; __syncthreads();
    for (int o = 128; o > 0; o >>= 1) {
        if (tid < o) red[tid] = fmaxf(red[tid], red[tid + o]);
        __syncthreads();
    }
    mx = red[0]; __syncthreads();

    float se = 0.0f;
    for (int n = tid; n < NN; n += 256) {
        float e = expf(alpha[n] - mx);
        alpha[n] = e;
        se += e;
    }
    red[tid] = se; __syncthreads();
    for (int o = 128; o > 0; o >>= 1) {
        if (tid < o) red[tid] += red[tid + o];
        __syncthreads();
    }
    float inv = 1.0f / red[0];
    __syncthreads();
    for (int n = tid; n < NN; n += 256) alpha[n] *= inv;
    __syncthreads();

    // ctx[d] = sum_n alpha[n] * features[b,n,d]   (features stays L2-resident)
    for (int d = tid; d < DV; d += 256) {
        float acc = 0.0f;
        for (int n = 0; n < NN; ++n)
            acc += alpha[n] * features[((size_t)b * NN + n) * DV + d];
        x[(size_t)b * XK + d] = (_Float16)acc;
    }
}

// emb_t gather -> x[:,512:1024] f16
__global__ void embed_gather(const float* __restrict__ emb_table,
                             const int* __restrict__ captions,
                             _Float16* __restrict__ x, int t)
{
    int i = blockIdx.x * blockDim.x + threadIdx.x;   // B*E
    if (i >= BB * EE) return;
    int b = i / EE, e = i % EE;
    int tok = captions[b * TT + t];
    x[(size_t)b * XK + DV + e] = (_Float16)emb_table[(size_t)tok * EE + e];
}

// h0 -> x[:,1024:2048] f16
__global__ void h_to_x(const float* __restrict__ hbuf, _Float16* __restrict__ x)
{
    int i = blockIdx.x * blockDim.x + threadIdx.x;   // B*H
    if (i >= BB * HH) return;
    int b = i / HH, j = i % HH;
    x[(size_t)b * XK + 1024 + j] = (_Float16)hbuf[i];
}

// LSTM pointwise; also writes h_new (f16) into x[:,1024:2048] for next GEMMs
__global__ void lstm_elem(const float* __restrict__ gates,
                          float* __restrict__ hbuf,
                          float* __restrict__ cbuf,
                          _Float16* __restrict__ x)
{
    int i = blockIdx.x * blockDim.x + threadIdx.x;   // B*H
    if (i >= BB * HH) return;
    int b = i / HH, j = i % HH;
    const float* g = gates + (size_t)b * G4H;
    float ig = g[j], fg = g[HH + j], gg = g[2 * HH + j], og = g[3 * HH + j];
    float si = 1.0f / (1.0f + expf(-ig));
    float sf = 1.0f / (1.0f + expf(-fg));
    float so = 1.0f / (1.0f + expf(-og));
    float cn = sf * cbuf[i] + si * tanhf(gg);
    float hn = so * tanhf(cn);
    cbuf[i] = cn;
    hbuf[i] = hn;
    x[(size_t)b * XK + 1024 + j] = (_Float16)hn;
}

// in-place per-row: read logits (in softmax half), write log_softmax + softmax
__global__ void softmax_rows(float* __restrict__ out_ls,
                             float* __restrict__ logits)
{
    int r = blockIdx.x;                      // T*B rows
    float* row = logits + (size_t)r * VV;
    float* lsr = out_ls + (size_t)r * VV;
    int tid = threadIdx.x;
    __shared__ float red[256];

    float mx = -3.402823e38f;
    for (int i = tid; i < VV; i += 256) mx = fmaxf(mx, row[i]);
    red[tid] = mx; __syncthreads();
    for (int o = 128; o > 0; o >>= 1) {
        if (tid < o) red[tid] = fmaxf(red[tid], red[tid + o]);
        __syncthreads();
    }
    mx = red[0]; __syncthreads();

    float se = 0.0f;
    for (int i = tid; i < VV; i += 256) se += expf(row[i] - mx);
    red[tid] = se; __syncthreads();
    for (int o = 128; o > 0; o >>= 1) {
        if (tid < o) red[tid] += red[tid + o];
        __syncthreads();
    }
    float sum = red[0];
    float lse = mx + logf(sum);
    float inv = 1.0f / sum;
    for (int i = tid; i < VV; i += 256) {
        float xv = row[i];
        lsr[i] = xv - lse;                 // log_softmax
        row[i] = expf(xv - mx) * inv;      // softmax (in place)
    }
}

// ---------------------------------------------------------------------------
extern "C" void kernel_launch(void* const* d_in, const int* in_sizes, int n_in,
                              void* d_out, int out_size, void* d_ws, size_t ws_size,
                              hipStream_t stream)
{
    const float* features   = (const float*)d_in[0];
    const int*   captions   = (const int*)  d_in[1];
    const float* W_init_h   = (const float*)d_in[2];
    const float* W_init_c   = (const float*)d_in[3];
    const float* W_attn_v   = (const float*)d_in[4];
    const float* b_attn_v   = (const float*)d_in[5];
    const float* W_attn_h   = (const float*)d_in[6];
    const float* b_attn_h   = (const float*)d_in[7];
    const float* emb_table  = (const float*)d_in[8];
    const float* W_ih       = (const float*)d_in[9];
    const float* W_hh       = (const float*)d_in[10];
    const float* b_ih       = (const float*)d_in[11];
    const float* b_hh       = (const float*)d_in[12];
    const float* W_out      = (const float*)d_in[13];
    const float* b_out      = (const float*)d_in[14];

    // workspace carve-out (256B aligned)
    char* ws = (char*)d_ws;
    size_t off = 0;
    auto carve = [&](size_t bytes) {
        char* p = ws + off;
        off += (bytes + 255) & ~(size_t)255;
        return p;
    };
    _Float16* Wcat_h   = (_Float16*)carve((size_t)G4H * XK * 2);   // 16 MB
    _Float16* Wout_h   = (_Float16*)carve((size_t)VV * HH * 2);    // 20 MB
    _Float16* Winh_h   = (_Float16*)carve((size_t)HH * DV * 2);
    _Float16* Winc_h   = (_Float16*)carve((size_t)HH * DV * 2);
    _Float16* fmean_h  = (_Float16*)carve((size_t)BB * DV * 2);
    _Float16* xbuf     = (_Float16*)carve((size_t)BB * XK * 2);
    float*    att_v    = (float*)   carve((size_t)BB * NN * 4);
    float*    hbuf     = (float*)   carve((size_t)BB * HH * 4);
    float*    cbuf     = (float*)   carve((size_t)BB * HH * 4);
    float*    gates    = (float*)   carve((size_t)BB * G4H * 4);
    float*    bias4h   = (float*)   carve((size_t)G4H * 4);

    float* out_ls = (float*)d_out;
    float* out_sm = out_ls + (size_t)TT * BB * VV;    // logits staged here, in place

    // GEMM grid helper: 4 waves/block, each wave MT x NT tiles
    auto gemm_grid = [](int M, int Nout) {
        int ntiles = (Nout + 15) / 16;
        int gx = (ntiles + 4 * NT - 1) / (4 * NT);
        int gy = (M / 16) / MT;
        return dim3((unsigned)gx, (unsigned)gy);
    };

    // ---- one-time conversions / precompute ----
    {
        int n = HH * DV;
        cvt_f32_f16<<<dim3((n + 255) / 256), dim3(256), 0, stream>>>(W_init_h, Winh_h, n);
        cvt_f32_f16<<<dim3((n + 255) / 256), dim3(256), 0, stream>>>(W_init_c, Winc_h, n);
    }
    {
        int n = G4H * XK;
        cvt_concat_wcat<<<dim3((n + 255) / 256), dim3(256), 0, stream>>>(W_ih, W_hh, Wcat_h);
    }
    {
        int n = VV * HH;
        cvt_f32_f16<<<dim3((n + 255) / 256), dim3(256), 0, stream>>>(W_out, Wout_h, n);
    }
    bias_combine<<<dim3((G4H + 255) / 256), dim3(256), 0, stream>>>(b_ih, b_hh, bias4h);
    featmean_kernel<<<dim3(BB), dim3(256), 0, stream>>>(features, fmean_h);
    attv_kernel<<<dim3(BB * NN), dim3(128), 0, stream>>>(features, W_attn_v, b_attn_v, att_v);

    // ---- h0 / c0 : (128x512) x (512x1024) WMMA GEMMs ----
    wmma_gemm_tiled<<<gemm_grid(BB, HH), dim3(128), 0, stream>>>(
        fmean_h, DV, Winh_h, DV, hbuf, HH, nullptr, BB, HH, DV);
    wmma_gemm_tiled<<<gemm_grid(BB, HH), dim3(128), 0, stream>>>(
        fmean_h, DV, Winc_h, DV, cbuf, HH, nullptr, BB, HH, DV);
    h_to_x<<<dim3(BB * HH / 256), dim3(256), 0, stream>>>(hbuf, xbuf);

    // ---- time loop ----
    for (int t = 0; t < TT; ++t) {
        attn_ctx_kernel<<<dim3(BB), dim3(256), 0, stream>>>(
            features, att_v, hbuf, W_attn_h, b_attn_h, xbuf);
        embed_gather<<<dim3(BB * EE / 256), dim3(256), 0, stream>>>(
            emb_table, captions, xbuf, t);

        // gates = [ctx|emb|h] @ [W_ih|W_hh]^T + (b_ih+b_hh)   K=2048, Nout=4096
        wmma_gemm_tiled<<<gemm_grid(BB, G4H), dim3(128), 0, stream>>>(
            xbuf, XK, Wcat_h, XK, gates, G4H, bias4h, BB, G4H, XK);

        lstm_elem<<<dim3(BB * HH / 256), dim3(256), 0, stream>>>(gates, hbuf, cbuf, xbuf);

        // logits_t = h_new @ W_out^T + b_out   K=1024, Nout=10000 (625 tiles)
        wmma_gemm_tiled<<<gemm_grid(BB, VV), dim3(128), 0, stream>>>(
            xbuf + 1024, XK, Wout_h, HH,
            out_sm + (size_t)t * BB * VV, VV, b_out, BB, VV, HH);
    }

    // ---- final packed log_softmax / softmax over (T*B, V) ----
    softmax_rows<<<dim3(TT * BB), dim3(256), 0, stream>>>(out_ls, out_sm);
}
